// EDESC_9921374454086
// MI455X (gfx1250) — matrix operand order repaired and checked by reading, not compiled, not run on hardware
//
#include <hip/hip_runtime.h>

// ---------------------------------------------------------------------------
// Types for CDNA5 WMMA (wave32): bf16 A/B fragments (16 bf16 per lane = 8 VGPR),
// f32 C/D accumulator (8 f32 per lane = 8 VGPR).
// ---------------------------------------------------------------------------
typedef __attribute__((ext_vector_type(8)))  __bf16 v8bf;
typedef __attribute__((ext_vector_type(16))) __bf16 v16bf;
typedef __attribute__((ext_vector_type(8)))  float  v8f;

union V16 { v16bf v; v8bf h[2]; };

#define TILE_N 128          // batch rows per block
#define TILE_M 128          // output cols per block
#define TILE_K 32           // K per WMMA step
#define LDSB   40           // LDS row stride in bf16 (32 + 8 pad -> no bank conflicts)
#define LDS_ELEMS (TILE_N * LDSB)

#define NROWS 16384

// ---------------------------------------------------------------------------
// fp32 W[K][M] -> bf16 Wt[M][K]  (transpose + convert; tiny one-shot cost)
// ---------------------------------------------------------------------------
__global__ void transpose_cvt_kernel(const float* __restrict__ W,
                                     __bf16* __restrict__ Wt, int K, int M) {
    size_t idx = (size_t)blockIdx.x * blockDim.x + threadIdx.x;
    size_t total = (size_t)K * M;
    if (idx < total) {
        int m = (int)(idx / K);
        int k = (int)(idx % K);
        Wt[idx] = (__bf16)W[(size_t)k * M + m];
    }
}

// ---------------------------------------------------------------------------
// Tiled bf16 WMMA GEMM with double-buffered LDS pipeline:
//   out[N][M] = act( A[N][K] @ Wt[M][K]^T + bias )
//   A_F32    : A is fp32 (first layer reads x directly), else bf16
//   RELU     : apply ReLU in epilogue
//   OUT_BF16 : write bf16 activation for the next layer
//   OUT_F32  : write fp32 result (final outputs / z / p)
// Block = 256 threads = 8 waves (2x4), wave tile = 64 rows x 32 cols
// = 4x2 v_wmma_f32_16x16x32_bf16 accumulators.
// Pipeline: fetch(k+1) -> compute(k) -> stage(k+1) -> barrier   (1 barrier/step,
// global-load latency hidden behind the WMMA phase).
// ---------------------------------------------------------------------------
template <bool A_F32, bool RELU, bool OUT_BF16, bool OUT_F32>
__global__ __launch_bounds__(256) void gemm_bf16_wmma(
    const void* __restrict__ Aptr, const __bf16* __restrict__ Wt,
    const float* __restrict__ bias, __bf16* __restrict__ outb,
    float* __restrict__ outf, int K, int M) {

    __shared__ __bf16 Alds[2][LDS_ELEMS];
    __shared__ __bf16 Blds[2][LDS_ELEMS];

    const int tid  = threadIdx.x;
    const int lane = tid & 31;
    const int wave = tid >> 5;
    const int wr   = wave >> 2;   // 0..1  (row half of block tile)
    const int wc   = wave & 3;    // 0..3  (col quarter of block tile)
    const int lm   = lane & 15;   // lane within 16-group
    const int lg   = lane >> 4;   // lane group (0/1)

    const size_t rowblk = (size_t)blockIdx.y * TILE_N;
    const size_t colblk = (size_t)blockIdx.x * TILE_M;

    // per-thread staging slots (2 chunks of 8 elems for each of A and B)
    const int c0r[2] = { tid >> 2, (tid + 256) >> 2 };
    const int c0c[2] = { (tid & 3) * 8, ((tid + 256) & 3) * 8 };

    const v8f vzero = {0.f, 0.f, 0.f, 0.f, 0.f, 0.f, 0.f, 0.f};
    v8f acc[4][2];
#pragma unroll
    for (int mi = 0; mi < 4; ++mi)
#pragma unroll
        for (int ni = 0; ni < 2; ++ni) acc[mi][ni] = vzero;

    // prefetch registers
    v8bf   ra[2], rb[2];
    float4 rfa[2][2];

    auto fetch = [&](int k0) {
#pragma unroll
        for (int i = 0; i < 2; ++i) {
            if (A_F32) {
                const float* src =
                    (const float*)Aptr + (rowblk + c0r[i]) * (size_t)K + k0 + c0c[i];
                rfa[i][0] = *(const float4*)(src);
                rfa[i][1] = *(const float4*)(src + 4);
            } else {
                ra[i] = *(const v8bf*)((const __bf16*)Aptr +
                                       (rowblk + c0r[i]) * (size_t)K + k0 + c0c[i]);
            }
            rb[i] = *(const v8bf*)(Wt + (colblk + c0r[i]) * (size_t)K + k0 + c0c[i]);
        }
    };
    auto stage = [&](int buf) {
#pragma unroll
        for (int i = 0; i < 2; ++i) {
            if (A_F32) {
                v8bf b;
                b[0] = (__bf16)rfa[i][0].x; b[1] = (__bf16)rfa[i][0].y;
                b[2] = (__bf16)rfa[i][0].z; b[3] = (__bf16)rfa[i][0].w;
                b[4] = (__bf16)rfa[i][1].x; b[5] = (__bf16)rfa[i][1].y;
                b[6] = (__bf16)rfa[i][1].z; b[7] = (__bf16)rfa[i][1].w;
                *(v8bf*)&Alds[buf][c0r[i] * LDSB + c0c[i]] = b;
            } else {
                *(v8bf*)&Alds[buf][c0r[i] * LDSB + c0c[i]] = ra[i];
            }
            *(v8bf*)&Blds[buf][c0r[i] * LDSB + c0c[i]] = rb[i];
        }
    };

    const int KT = K / TILE_K;

    fetch(0);
    stage(0);
    __syncthreads();

    for (int kt = 0; kt < KT; ++kt) {
        const int cur = kt & 1;
        if (kt + 1 < KT) fetch((kt + 1) * TILE_K);   // loads in flight over compute

        // ---- fragments (ISA 16-bit layouts) ----
        // A 16x32: lane group g holds K = {8g..8g+7} U {16+8g..16+8g+7}
        V16 af[4];
#pragma unroll
        for (int mi = 0; mi < 4; ++mi) {
            int r  = wr * 64 + mi * 16 + lm;
            int hk = lg * 8;
            af[mi].h[0] = *(const v8bf*)&Alds[cur][r * LDSB + hk];
            af[mi].h[1] = *(const v8bf*)&Alds[cur][r * LDSB + hk + 16];
        }
        // B 32x16: lane group g holds K = {8g..8g+7} U {16+8g..16+8g+7} for col lm
        V16 bfr[2];
#pragma unroll
        for (int ni = 0; ni < 2; ++ni) {
            int n  = wc * 32 + ni * 16 + lm;
            int kk = lg * 8;
            bfr[ni].h[0] = *(const v8bf*)&Blds[cur][n * LDSB + kk];
            bfr[ni].h[1] = *(const v8bf*)&Blds[cur][n * LDSB + kk + 16];
        }
#pragma unroll
        for (int mi = 0; mi < 4; ++mi)
#pragma unroll
            for (int ni = 0; ni < 2; ++ni)
                acc[mi][ni] = __builtin_amdgcn_wmma_f32_16x16x32_bf16(
                    false, af[mi].v, false, bfr[ni].v, (short)0, acc[mi][ni],
                    false, false);

        if (kt + 1 < KT) stage((kt + 1) & 1);        // waits on loads only here
        __syncthreads();
    }

    // ---- epilogue: C/D layout -> col = lane&15, row = vgpr + 8*(lane>>4)
#pragma unroll
    for (int ni = 0; ni < 2; ++ni) {
        size_t j  = colblk + wc * 32 + ni * 16 + lm;
        float  bb = bias ? bias[j] : 0.0f;
#pragma unroll
        for (int mi = 0; mi < 4; ++mi) {
            v8f c = acc[mi][ni];
#pragma unroll
            for (int v = 0; v < 8; ++v) {
                size_t row = rowblk + wr * 64 + mi * 16 + v + 8 * lg;
                float  val = c[v] + bb;
                if (RELU) val = fmaxf(val, 0.0f);
                if (OUT_F32)  outf[row * (size_t)M + j] = val;
                if (OUT_BF16) outb[row * (size_t)M + j] = (__bf16)val;
            }
        }
    }
}

// ---------------------------------------------------------------------------
// Subspace affinity: p[N][256] -> s[N][32]
// one wave per row; lane = cluster; 8-elem sum of squares, affine, row-normalize
// ---------------------------------------------------------------------------
__global__ __launch_bounds__(256) void affinity_kernel(const float* __restrict__ p,
                                                       float* __restrict__ sout) {
    const int wave = threadIdx.x >> 5;
    const int lane = threadIdx.x & 31;
    const size_t row = (size_t)blockIdx.x * 8 + wave;
    const float* pr = p + row * 256 + lane * 8;
    float4 a = *(const float4*)(pr);
    float4 b = *(const float4*)(pr + 4);
    float s = a.x * a.x + a.y * a.y + a.z * a.z + a.w * a.w +
              b.x * b.x + b.y * b.y + b.z * b.z + b.w * b.w;
    // (s + ETA*D_SUB) / ((ETA+1)*D_SUB), ETA=5, D_SUB=8
    s = (s + 40.0f) * (1.0f / 48.0f);
    float tot = s;
#pragma unroll
    for (int off = 16; off > 0; off >>= 1) tot += __shfl_xor(tot, off, 32);
    sout[row * 32 + lane] = s / tot;
}

// ---------------------------------------------------------------------------
// Host orchestration
// ---------------------------------------------------------------------------
extern "C" void kernel_launch(void* const* d_in, const int* in_sizes, int n_in,
                              void* d_out, int out_size, void* d_ws, size_t ws_size,
                              hipStream_t stream) {
    (void)in_sizes; (void)n_in; (void)out_size; (void)ws_size;

    const float* x = (const float*)d_in[0];
    const float* W[9] = {
        (const float*)d_in[1],  (const float*)d_in[3],  (const float*)d_in[5],
        (const float*)d_in[7],  (const float*)d_in[9],  (const float*)d_in[11],
        (const float*)d_in[13], (const float*)d_in[15], (const float*)d_in[17] };
    const float* B[8] = {
        (const float*)d_in[2],  (const float*)d_in[4],  (const float*)d_in[6],
        (const float*)d_in[8],  (const float*)d_in[10], (const float*)d_in[12],
        (const float*)d_in[14], (const float*)d_in[16] };
    // (K, M) per weight, in reference order (last = D: 128 x 256)
    const int KM[9][2] = { {2048,1024},{1024,1024},{1024,2048},{2048,128},
                           {128,2048},{2048,1024},{1024,1024},{1024,2048},{128,256} };

    // Outputs, concatenated flat: x_bar [N,2048], s [N,32], z [N,128]
    float* out_xbar = (float*)d_out;
    float* out_s    = out_xbar + (size_t)NROWS * 2048;
    float* out_z    = out_s    + (size_t)NROWS * 32;

    // Workspace carve (all regions 16B aligned)
    __bf16* actA = (__bf16*)d_ws;                       // 16384*2048 bf16
    __bf16* actB = actA + (size_t)NROWS * 2048;         // 16384*2048 bf16
    __bf16* zb   = actB + (size_t)NROWS * 2048;         // 16384*128  bf16
    __bf16* wtb  = zb   + (size_t)NROWS * 128;          // bf16 weights (transposed)
    __bf16* Wt[9];
    size_t off = 0;
    for (int i = 0; i < 9; ++i) { Wt[i] = wtb + off; off += (size_t)KM[i][0] * KM[i][1]; }
    float* p = (float*)(wtb + off);                     // 16384*256 fp32

    // 1) convert+transpose all weights to bf16 Wt[M][K]
    for (int i = 0; i < 9; ++i) {
        size_t total = (size_t)KM[i][0] * KM[i][1];
        transpose_cvt_kernel<<<(unsigned)((total + 255) / 256), 256, 0, stream>>>(
            W[i], Wt[i], KM[i][0], KM[i][1]);
    }

    const dim3 blk(256);
    auto grd = [](int M) { return dim3((unsigned)(M / TILE_M), NROWS / TILE_N); };

    // 2) encoder
    gemm_bf16_wmma<true,  true,  true,  false><<<grd(1024), blk, 0, stream>>>(
        x,    Wt[0], B[0], actA, nullptr, 2048, 1024);                 // h1
    gemm_bf16_wmma<false, true,  true,  false><<<grd(1024), blk, 0, stream>>>(
        actA, Wt[1], B[1], actB, nullptr, 1024, 1024);                 // h2
    gemm_bf16_wmma<false, true,  true,  false><<<grd(2048), blk, 0, stream>>>(
        actB, Wt[2], B[2], actA, nullptr, 1024, 2048);                 // h3
    gemm_bf16_wmma<false, false, true,  true ><<<grd(128),  blk, 0, stream>>>(
        actA, Wt[3], B[3], zb,   out_z,  2048, 128);                   // z (bf16 + f32)
    // 3) decoder
    gemm_bf16_wmma<false, true,  true,  false><<<grd(2048), blk, 0, stream>>>(
        zb,   Wt[4], B[4], actB, nullptr, 128,  2048);                 // h4
    gemm_bf16_wmma<false, true,  true,  false><<<grd(1024), blk, 0, stream>>>(
        actB, Wt[5], B[5], actA, nullptr, 2048, 1024);                 // h5
    gemm_bf16_wmma<false, true,  true,  false><<<grd(1024), blk, 0, stream>>>(
        actA, Wt[6], B[6], actB, nullptr, 1024, 1024);                 // h6
    gemm_bf16_wmma<false, false, false, true ><<<grd(2048), blk, 0, stream>>>(
        actB, Wt[7], B[7], nullptr, out_xbar, 1024, 2048);             // x_bar
    // 4) affinity: p = z @ D, then s
    gemm_bf16_wmma<false, false, false, true ><<<grd(256),  blk, 0, stream>>>(
        zb,   Wt[8], nullptr, nullptr, p, 128, 256);                   // p
    affinity_kernel<<<NROWS / 8, blk, 0, stream>>>(p, out_s);
}
